// ASSA_32830730010703
// MI455X (gfx1250) — compile-verified, matrix operand order checked by reading, not run on hardware
//
#include <hip/hip_runtime.h>
#include <cstdint>

// ---------------------------------------------------------------------------
// ASSA dual-branch attention for MI455X (gfx1250, wave32, WMMA bf16, TDM)
// ---------------------------------------------------------------------------

#define DEV __device__ __forceinline__

typedef __attribute__((ext_vector_type(16))) __bf16 v16bf;
typedef __attribute__((ext_vector_type(8)))  __bf16 v8bf;
typedef __attribute__((ext_vector_type(8)))  float  v8f;
typedef __attribute__((ext_vector_type(4)))  unsigned int u32x4;
typedef __attribute__((ext_vector_type(8)))  int i32x8;
typedef __attribute__((ext_vector_type(4)))  int i32x4;

static constexpr int B_    = 8;
static constexpr int N_    = 1024;     // H*W
static constexpr int C_    = 512;
static constexpr int HD    = 64;       // head dim
static constexpr int BH    = B_ * 8;              // 64 (batch*heads)
static constexpr int TOK   = B_ * N_;             // 8192
static constexpr float SCALE = 0.125f;            // 64^-0.5
static constexpr float EPS   = 1e-3f;

#if defined(__gfx1250__) && __has_builtin(__builtin_amdgcn_tensor_load_to_lds)
#define HAVE_TDM 1
#else
#define HAVE_TDM 0
#endif
#if defined(__gfx1250__) && __has_builtin(__builtin_amdgcn_global_load_async_to_lds_b128)
#define HAVE_ASYNC 1
#else
#define HAVE_ASYNC 0
#endif

#define AS1 __attribute__((address_space(1)))
#define AS3 __attribute__((address_space(3)))

DEV v8f wmma_bf16(v16bf a, v16bf b, v8f c) {
    return __builtin_amdgcn_wmma_f32_16x16x32_bf16(false, a, false, b,
                                                   (short)0, c, false, false);
}

// Load one 16x32 bf16 A/B fragment from a row-major tile.
// Per-lane layout (CDNA5 ISA 7.12.2): lane l, r=l&15, h=l>>4 holds
// row r, K = [8h .. 8h+7] and [16+8h .. 16+8h+7].
DEV v16bf ld_frag(const __bf16* p, int pitch) {
    const int l = threadIdx.x & 31, r = l & 15, h = l >> 4;
    const __bf16* q = p + (size_t)r * pitch + h * 8;
    v8bf lo = *(const v8bf*)q;
    v8bf hi = *(const v8bf*)(q + 16);
    return __builtin_shufflevector(lo, hi, 0,1,2,3,4,5,6,7,8,9,10,11,12,13,14,15);
}

DEV float wave_sum16(float v) {      // reduce across the 16 lanes of a half
    #pragma unroll
    for (int m = 8; m >= 1; m >>= 1) v += __shfl_xor(v, m, 32);
    return v;
}
DEV float wave_max16(float v) {
    #pragma unroll
    for (int m = 8; m >= 1; m >>= 1) v = fmaxf(v, __shfl_xor(v, m, 32));
    return v;
}

// ------------------ CDNA5 async/TDM tile movers ----------------------------
DEV void wait_async0() {
#if defined(__gfx1250__)
#if __has_builtin(__builtin_amdgcn_s_wait_asynccnt)
    __builtin_amdgcn_s_wait_asynccnt(0);
#else
    asm volatile("s_wait_asynccnt 0" ::: "memory");
#endif
#endif
}

// 16-byte global->LDS copy chunk (async path if the toolchain exposes it).
DEV void cp_lds_16B(__bf16* dst, const __bf16* src) {
#if HAVE_ASYNC
    __builtin_amdgcn_global_load_async_to_lds_b128(
        (AS1 i32x4*)(uintptr_t)src,
        (AS3 i32x4*)(unsigned)(uintptr_t)dst, 0, 0);
#else
    *(v8bf*)dst = *(const v8bf*)src;
#endif
}

#if HAVE_TDM
// 2-D TDM tile load: tile (tile_w x tile_h) elems of 2B out of a tensor with
// row length tensor_w, tensor_h rows, dim0 stride `stride` (elements).
DEV void tdm_load_2d(const void* lds, const void* glb,
                     unsigned tile_w, unsigned tile_h,
                     unsigned tensor_w, unsigned tensor_h, unsigned stride) {
    unsigned long long ga = (unsigned long long)(uintptr_t)glb;
    u32x4 g0 = { 1u,                                    // count=1, user mode
                 (unsigned)(uintptr_t)lds,              // lds_addr
                 (unsigned)ga,                          // global_addr[95:64]
                 (unsigned)((ga >> 32) & 0x1ffffffu) | (2u << 30) }; // type=2
    i32x8 g1 = { (int)(1u << 16),                       // data_size = 2B
                 (int)((tensor_w & 0xffffu) << 16),     // tensor_dim0[15:0]
                 (int)((tensor_w >> 16) | ((tensor_h & 0xffffu) << 16)),
                 (int)((tensor_h >> 16) | (tile_w << 16)),   // tile_dim0
                 (int)(tile_h & 0xffffu),               // tile_dim1
                 (int)stride,                           // tensor_dim0_stride
                 0, 0 };
    i32x4 z4 = { 0, 0, 0, 0 };
    i32x8 z8 = { 0, 0, 0, 0, 0, 0, 0, 0 };
    __builtin_amdgcn_tensor_load_to_lds(g0, g1, z4, z4, z8, 0);
}
#endif

// ------------------------------- LayerNorm ---------------------------------
__global__ __launch_bounds__(256)
void assa_layernorm(const float* __restrict__ x, const float* __restrict__ g,
                    const float* __restrict__ bta, __bf16* __restrict__ out) {
    const int tok = blockIdx.x, t = threadIdx.x;
    __shared__ float red[2][8];
    __shared__ float stats[2];
    float2 v = ((const float2*)x)[(size_t)tok * (C_ / 2) + t];
    float s  = v.x + v.y;
    float sq = v.x * v.x + v.y * v.y;
    #pragma unroll
    for (int m = 16; m >= 1; m >>= 1) { s += __shfl_xor(s, m, 32); sq += __shfl_xor(sq, m, 32); }
    if ((t & 31) == 0) { red[0][t >> 5] = s; red[1][t >> 5] = sq; }
    __syncthreads();
    if (t == 0) {
        float ts = 0.f, tq = 0.f;
        #pragma unroll
        for (int i = 0; i < 8; i++) { ts += red[0][i]; tq += red[1][i]; }
        float mu  = ts * (1.0f / C_);
        float var = tq * (1.0f / C_) - mu * mu;
        stats[0] = mu; stats[1] = __frsqrt_rn(var + EPS);
    }
    __syncthreads();
    const float mu = stats[0], ri = stats[1];
    float y0 = (v.x - mu) * ri * g[2 * t]     + bta[2 * t];
    float y1 = (v.y - mu) * ri * g[2 * t + 1] + bta[2 * t + 1];
    __bf16* o = out + (size_t)tok * C_ + 2 * t;
    o[0] = (__bf16)y0; o[1] = (__bf16)y1;
}

// ------------------ weight transpose f32 [K,N] -> bf16 [N,K] ---------------
__global__ __launch_bounds__(256)
void assa_wtrans(const float* __restrict__ in, __bf16* __restrict__ out,
                 int K, int Nc) {
    __shared__ float tile[32][33];
    const int n0 = blockIdx.x * 32, k0 = blockIdx.y * 32;
    const int tx = threadIdx.x, ty = threadIdx.y;   // (32, 8)
    #pragma unroll
    for (int i = 0; i < 4; i++)
        tile[ty + 8 * i][tx] = in[(size_t)(k0 + ty + 8 * i) * Nc + n0 + tx];
    __syncthreads();
    #pragma unroll
    for (int i = 0; i < 4; i++)
        out[(size_t)(n0 + ty + 8 * i) * K + k0 + tx] = (__bf16)tile[tx][ty + 8 * i];
}

// --------------------- QKV GEMM + scatter/activation epilogue --------------
__global__ __launch_bounds__(256)
void assa_qkv_gemm(const __bf16* __restrict__ X, const __bf16* __restrict__ Wt,
                   const float* __restrict__ bias,
                   __bf16* __restrict__ qo, __bf16* __restrict__ ko,
                   __bf16* __restrict__ vto, int sparse) {
    const int w  = threadIdx.x >> 5;
    const int r0 = blockIdx.x * 128 + w * 16;       // token rows
    const int c0 = blockIdx.y * 64;                 // output cols
    v8f acc[4] = {};
    for (int k = 0; k < C_; k += 32) {
        if (k + 32 < C_)
            __builtin_prefetch(Wt + (size_t)c0 * C_ + k + 32, 0, 1);
        v16bf a = ld_frag(X + (size_t)r0 * C_ + k, C_);
        #pragma unroll
        for (int cf = 0; cf < 4; cf++) {
            v16bf b = ld_frag(Wt + (size_t)(c0 + cf * 16) * C_ + k, C_);
            acc[cf] = wmma_bf16(a, b, acc[cf]);
        }
    }
    const int l = threadIdx.x & 31, r = l & 15, h = l >> 4;
    #pragma unroll
    for (int cf = 0; cf < 4; cf++) {
        #pragma unroll
        for (int j = 0; j < 8; j++) {
            const int row = r0 + j + 8 * h;         // token
            const int c   = c0 + cf * 16 + r;       // 0..1535
            float y = acc[cf][j] + bias[c];
            const int part = c >> 9, ch = (c >> 6) & 7, dd = c & 63;
            const int b = row >> 10, n = row & 1023;
            const int bh = b * 8 + ch;
            if (sparse && part < 2) { y = fmaxf(y, 0.f); y = y * y; }
            const __bf16 v = (__bf16)y;
            if (part == 0) {
                qo[((size_t)bh * N_ + n) * HD + dd] = v;
            } else if (part == 1) {
                if (sparse) ko[((size_t)bh * HD + dd) * N_ + n] = v;   // rk^T
                else        ko[((size_t)bh * N_ + n) * HD + dd] = v;   // row-major
            } else {
                vto[((size_t)bh * HD + dd) * N_ + n] = v;              // v^T
            }
        }
    }
}

// -------------------------- dense flash attention --------------------------
// grid (N/128, BH); block 256 = 8 waves; wave handles 16 query rows.
__global__ __launch_bounds__(256)
void assa_flash_dense(const __bf16* __restrict__ qd, const __bf16* __restrict__ kd,
                      const __bf16* __restrict__ vdt, __bf16* __restrict__ outd) {
    __shared__ __bf16 ksh[64 * HD];           // 64 keys, row-major [m][d]
    __shared__ __bf16 vsh[HD * 64];           // v^T  [d][m]
    __shared__ __bf16 psh[8 * 16 * 64];       // per-wave P tiles
    const int bh = blockIdx.y;
    const int w  = threadIdx.x >> 5;
    const int qr0 = blockIdx.x * 128 + w * 16;
    const int l = threadIdx.x & 31, r = l & 15, h = l >> 4;
    const size_t base = (size_t)bh << 16;     // bh * 1024 * 64

    v16bf qf[2];
    #pragma unroll
    for (int t = 0; t < 2; t++)
        qf[t] = ld_frag(qd + base + (size_t)qr0 * HD + t * 32, HD);

    v8f O[4] = {};
    float mi[8], li[8];
    #pragma unroll
    for (int j = 0; j < 8; j++) { mi[j] = -1e30f; li[j] = 0.f; }

    __bf16* pw = psh + w * 16 * 64;

    for (int mt = 0; mt < N_ / 64; mt++) {
        __syncthreads();                      // previous tile fully consumed
#if HAVE_TDM
        if (w == 0) {
            // K tile: contiguous 64x64 bf16 (1-D, 4096 elems)
            tdm_load_2d(ksh, kd + base + (size_t)mt * 64 * HD,
                        4096u, 1u, 4096u, 1u, 4096u);
            // V^T tile: 64 rows x 64 elems out of [64][1024] tensor
            tdm_load_2d(vsh, vdt + base + (size_t)mt * 64,
                        64u, 64u, 1024u, 64u, 1024u);
            __builtin_amdgcn_s_wait_tensorcnt(0);
        }
#else
        {   // async (or plain) 16B-chunk staging
            for (int i = threadIdx.x; i < 512; i += 256)
                cp_lds_16B(&ksh[i * 8], kd + base + (size_t)mt * 64 * HD + i * 8);
            for (int i = threadIdx.x; i < 512; i += 256) {
                const int d = i >> 3, c = (i & 7) * 8;
                cp_lds_16B(&vsh[d * 64 + c],
                           vdt + base + (size_t)d * N_ + mt * 64 + c);
            }
            wait_async0();
        }
#endif
        __syncthreads();

        // S = scale * Q K^T   (16 x 64)
        v8f S[4];
        #pragma unroll
        for (int mf = 0; mf < 4; mf++) {
            v8f s = {};
            #pragma unroll
            for (int t = 0; t < 2; t++)
                s = wmma_bf16(qf[t], ld_frag(ksh + (mf * 16) * HD + t * 32, HD), s);
            #pragma unroll
            for (int j = 0; j < 8; j++) S[mf][j] = s[j] * SCALE;
        }

        // online softmax update per row
        #pragma unroll
        for (int j = 0; j < 8; j++) {
            float cm = fmaxf(fmaxf(S[0][j], S[1][j]), fmaxf(S[2][j], S[3][j]));
            cm = wave_max16(cm);
            const float nm = fmaxf(mi[j], cm);
            const float corr = __expf(mi[j] - nm);
            mi[j] = nm;
            float rs = 0.f;
            #pragma unroll
            for (int mf = 0; mf < 4; mf++) {
                const float p = __expf(S[mf][j] - nm);
                S[mf][j] = p; rs += p;
            }
            rs = wave_sum16(rs);
            li[j] = li[j] * corr + rs;
            #pragma unroll
            for (int df = 0; df < 4; df++) O[df][j] *= corr;
        }

        // P -> bf16 A-layout via wave-private LDS
        #pragma unroll
        for (int mf = 0; mf < 4; mf++)
            #pragma unroll
            for (int j = 0; j < 8; j++)
                pw[(j + 8 * h) * 64 + mf * 16 + r] = (__bf16)S[mf][j];

        // O += P @ V
        #pragma unroll
        for (int t = 0; t < 2; t++) {
            v16bf a = ld_frag(pw + t * 32, 64);
            #pragma unroll
            for (int df = 0; df < 4; df++)
                O[df] = wmma_bf16(a, ld_frag(vsh + (df * 16) * 64 + t * 32, 64), O[df]);
        }
    }

    // normalize + store [bh][n][d] bf16
    #pragma unroll
    for (int j = 0; j < 8; j++) {
        const float inv = 1.f / li[j];
        const int row = qr0 + j + 8 * h;
        #pragma unroll
        for (int df = 0; df < 4; df++)
            outd[base + (size_t)row * HD + df * 16 + r] = (__bf16)(O[df][j] * inv);
    }
}

// -------------------- sparse branch: M2 = scale * rk^T @ v -----------------
__global__ __launch_bounds__(128)
void assa_sparse_m2(const __bf16* __restrict__ rkt, const __bf16* __restrict__ vst,
                    __bf16* __restrict__ m2t) {
    const int bh = blockIdx.x, w = threadIdx.x >> 5;
    const int r0 = w * 16;
    const size_t base = (size_t)bh << 16;
    v8f acc[4] = {};
    for (int m = 0; m < N_; m += 32) {
        v16bf a = ld_frag(rkt + base + (size_t)r0 * N_ + m, N_);
        #pragma unroll
        for (int cf = 0; cf < 4; cf++)
            acc[cf] = wmma_bf16(a, ld_frag(vst + base + (size_t)(cf * 16) * N_ + m, N_), acc[cf]);
    }
    const int l = threadIdx.x & 31, r = l & 15, h = l >> 4;
    #pragma unroll
    for (int cf = 0; cf < 4; cf++)
        #pragma unroll
        for (int j = 0; j < 8; j++) {
            const int d = r0 + j + 8 * h, d2 = cf * 16 + r;
            m2t[(size_t)bh * 4096 + d2 * 64 + d] = (__bf16)(acc[cf][j] * SCALE);
        }
}

// -------------------- sparse branch: out_s = rq @ M2 -----------------------
__global__ __launch_bounds__(256)
void assa_sparse_out(const __bf16* __restrict__ rq, const __bf16* __restrict__ m2t,
                     __bf16* __restrict__ outs) {
    const int bh = blockIdx.y, w = threadIdx.x >> 5;
    const int r0 = blockIdx.x * 128 + w * 16;
    const size_t base = (size_t)bh << 16;
    const __bf16* m2 = m2t + (size_t)bh * 4096;
    v8f acc[4] = {};
    #pragma unroll
    for (int t = 0; t < 2; t++) {
        v16bf a = ld_frag(rq + base + (size_t)r0 * HD + t * 32, HD);
        #pragma unroll
        for (int cf = 0; cf < 4; cf++)
            acc[cf] = wmma_bf16(a, ld_frag(m2 + (cf * 16) * 64 + t * 32, 64), acc[cf]);
    }
    const int l = threadIdx.x & 31, r = l & 15, h = l >> 4;
    #pragma unroll
    for (int cf = 0; cf < 4; cf++)
        #pragma unroll
        for (int j = 0; j < 8; j++)
            outs[base + (size_t)(r0 + j + 8 * h) * HD + cf * 16 + r] = (__bf16)acc[cf][j];
}

// ------------------ projection (both branches) + alpha fuse ----------------
DEV v16bf ld_frag_heads(const __bf16* b0, int r0, int k) {
    const int l = threadIdx.x & 31, r = l & 15, h = l >> 4;
    const int tok = r0 + r, b = tok >> 10, n = tok & 1023;
    const __bf16* p = b0 + ((size_t)(b * 8 + (k >> 6)) * N_ + n) * HD + (k & 63) + h * 8;
    v8bf lo = *(const v8bf*)p;
    v8bf hi = *(const v8bf*)(p + 16);
    return __builtin_shufflevector(lo, hi, 0,1,2,3,4,5,6,7,8,9,10,11,12,13,14,15);
}

__global__ __launch_bounds__(256)
void assa_proj_fuse(const __bf16* __restrict__ ad, const __bf16* __restrict__ as,
                    const __bf16* __restrict__ wdt, const __bf16* __restrict__ wst,
                    const float* __restrict__ bd, const float* __restrict__ bs,
                    const float* __restrict__ fw, float* __restrict__ out) {
    const int w  = threadIdx.x >> 5;
    const int r0 = blockIdx.x * 128 + w * 16;
    const int c0 = blockIdx.y * 64;
    v8f accD[4] = {}, accS[4] = {};
    for (int k = 0; k < C_; k += 32) {
        if (k + 32 < C_) {
            __builtin_prefetch(wdt + (size_t)c0 * C_ + k + 32, 0, 1);
            __builtin_prefetch(wst + (size_t)c0 * C_ + k + 32, 0, 1);
        }
        v16bf aD = ld_frag_heads(ad, r0, k);
        v16bf aS = ld_frag_heads(as, r0, k);
        #pragma unroll
        for (int cf = 0; cf < 4; cf++) {
            accD[cf] = wmma_bf16(aD, ld_frag(wdt + (size_t)(c0 + cf * 16) * C_ + k, C_), accD[cf]);
            accS[cf] = wmma_bf16(aS, ld_frag(wst + (size_t)(c0 + cf * 16) * C_ + k, C_), accS[cf]);
        }
    }
    const float e0 = __expf(fw[0]), e1 = __expf(fw[1]);
    const float aS_ = e0 / (e0 + e1), aD_ = e1 / (e0 + e1);
    const int l = threadIdx.x & 31, r = l & 15, h = l >> 4;
    #pragma unroll
    for (int cf = 0; cf < 4; cf++)
        #pragma unroll
        for (int j = 0; j < 8; j++) {
            const int tok = r0 + j + 8 * h, c = c0 + cf * 16 + r;
            out[(size_t)tok * C_ + c] =
                aS_ * (accS[cf][j] + bs[c]) + aD_ * (accD[cf][j] + bd[c]);
        }
}

// ---------------------------------------------------------------------------
extern "C" void kernel_launch(void* const* d_in, const int* in_sizes, int n_in,
                              void* d_out, int out_size, void* d_ws, size_t ws_size,
                              hipStream_t stream) {
    const float* x      = (const float*)d_in[0];
    const float* lng    = (const float*)d_in[1];
    const float* lnb    = (const float*)d_in[2];
    const float* qkvdw  = (const float*)d_in[3];
    const float* qkvdb  = (const float*)d_in[4];
    const float* qkvsw  = (const float*)d_in[5];
    const float* qkvsb  = (const float*)d_in[6];
    const float* projdw = (const float*)d_in[7];
    const float* projdb = (const float*)d_in[8];
    const float* projsw = (const float*)d_in[9];
    const float* projsb = (const float*)d_in[10];
    const float* fw     = (const float*)d_in[11];
    float* out = (float*)d_out;

    char* w = (char*)d_ws;
    auto alloc = [&](size_t elems) {
        void* p = (void*)w;
        w += (elems * sizeof(__bf16) + 255) & ~size_t(255);
        return (__bf16*)p;
    };
    __bf16* xln   = alloc((size_t)TOK * C_);
    __bf16* wqd   = alloc((size_t)3 * C_ * C_);
    __bf16* wqs   = alloc((size_t)3 * C_ * C_);
    __bf16* wpd   = alloc((size_t)C_ * C_);
    __bf16* wps   = alloc((size_t)C_ * C_);
    __bf16* q_d   = alloc((size_t)BH * N_ * HD);
    __bf16* k_d   = alloc((size_t)BH * N_ * HD);
    __bf16* v_dt  = alloc((size_t)BH * N_ * HD);
    __bf16* rq    = alloc((size_t)BH * N_ * HD);
    __bf16* rk_t  = alloc((size_t)BH * N_ * HD);
    __bf16* v_st  = alloc((size_t)BH * N_ * HD);
    __bf16* attnd = alloc((size_t)BH * N_ * HD);
    __bf16* attns = alloc((size_t)BH * N_ * HD);
    __bf16* m2t   = alloc((size_t)BH * HD * HD);

    assa_layernorm<<<TOK, 256, 0, stream>>>(x, lng, lnb, xln);

    dim3 tb(32, 8);
    assa_wtrans<<<dim3(48, 16), tb, 0, stream>>>(qkvdw,  wqd, C_, 3 * C_);
    assa_wtrans<<<dim3(48, 16), tb, 0, stream>>>(qkvsw,  wqs, C_, 3 * C_);
    assa_wtrans<<<dim3(16, 16), tb, 0, stream>>>(projdw, wpd, C_, C_);
    assa_wtrans<<<dim3(16, 16), tb, 0, stream>>>(projsw, wps, C_, C_);

    assa_qkv_gemm<<<dim3(TOK / 128, 24), 256, 0, stream>>>(xln, wqd, qkvdb,
                                                           q_d, k_d, v_dt, 0);
    assa_qkv_gemm<<<dim3(TOK / 128, 24), 256, 0, stream>>>(xln, wqs, qkvsb,
                                                           rq, rk_t, v_st, 1);

    assa_flash_dense<<<dim3(N_ / 128, BH), 256, 0, stream>>>(q_d, k_d, v_dt, attnd);

    assa_sparse_m2 <<<BH, 128, 0, stream>>>(rk_t, v_st, m2t);
    assa_sparse_out<<<dim3(N_ / 128, BH), 256, 0, stream>>>(rq, m2t, attns);

    assa_proj_fuse<<<dim3(TOK / 128, C_ / 64), 256, 0, stream>>>(
        attnd, attns, wpd, wps, projdb, projsb, fw, out);
}
